// SelfAttention_36318243455175
// MI455X (gfx1250) — matrix operand order repaired and checked by reading, not compiled, not run on hardware
//
#include <hip/hip_runtime.h>
#include <hip/hip_bf16.h>
#include <math.h>

// ---------------- types ----------------
typedef _Float16 v16h __attribute__((ext_vector_type(16)));
typedef _Float16 v8h  __attribute__((ext_vector_type(8)));
typedef float    v8f  __attribute__((ext_vector_type(8)));
typedef float    v4f  __attribute__((ext_vector_type(4)));

// ---------------- problem dims ----------------
#define BQ   8      // batch
#define CC   512    // channels
#define LL   1024   // H*W tokens
#define HPQ  1536   // 3*C
#define LC   77     // cond length
#define LCP  80     // padded cond length
#define DD   768    // cond dim
#define NH   8      // heads
#define CHD  64     // head dim

// ================= WMMA fragment helpers =================
// A 16x32 f16 (row-major [M][K] tile in LDS): lane&15=M row; half selects K
// interleave per ISA table: e<8 -> K=8h+e ; e>=8 -> K=16+8h+(e-8)
static __device__ __forceinline__ v16h frag_a_rm(const _Float16* __restrict__ p,
                                                 int ld, int lane) {
  const int m = lane & 15, hh = lane >> 4;
  const _Float16* r = p + (size_t)m * ld + hh * 8;
  v8h lo = *(const v8h*)(r);
  v8h hi = *(const v8h*)(r + 16);
  return __builtin_shufflevector(lo, hi, 0,1,2,3,4,5,6,7,8,9,10,11,12,13,14,15);
}
// B 32x16 f16 from an N-major (transposed) tile [N][Kld]:
// lane&15 = N column, K interleave per ISA table: e<8 -> K=8h+e ; e>=8 -> K=16+8h+(e-8)
static __device__ __forceinline__ v16h frag_b_nm(const _Float16* __restrict__ p,
                                                 int ldk, int lane) {
  const _Float16* r = p + (size_t)(lane & 15) * ldk + (lane >> 4) * 8;
  v8h lo = *(const v8h*)(r);
  v8h hi = *(const v8h*)(r + 16);
  return __builtin_shufflevector(lo, hi, 0,1,2,3,4,5,6,7,8,9,10,11,12,13,14,15);
}
static __device__ __forceinline__ v8f wmma16(v16h a, v16h b, v8f c) {
  return __builtin_amdgcn_wmma_f32_16x16x32_f16(false, a, false, b, (short)0, c,
                                                false, false);
}
static __device__ __forceinline__ float gelu_exact(float v) {
  return 0.5f * v * (1.f + erff(v * 0.7071067811865476f));
}
#define SHFLX(v, m) __shfl_xor((v), (m), 32)

// ================= utility: zero f16 buffer =================
__global__ __launch_bounds__(256) void zero_f16_kernel(_Float16* p, int n) {
  int i = blockIdx.x * 256 + threadIdx.x;
  if (i < n) p[i] = (_Float16)0.f;
}

// ================= GroupNorm (32 groups) -> f16 =================
// one block per (batch, group): 16 channels x 1024; channel-major in/out
__global__ __launch_bounds__(256) void groupnorm_kernel(
    const float* __restrict__ x, const float* __restrict__ gw,
    const float* __restrict__ gb, _Float16* __restrict__ outH) {
  const int g = blockIdx.x & 31, n = blockIdx.x >> 5;
  const int tid = threadIdx.x, lane = tid & 31, w = tid >> 5;
  const size_t base = ((size_t)n * CC + g * 16) * LL;
  const float* xp = x + base;
  float s = 0.f, ss = 0.f;
  for (int i = tid; i < 16 * LL; i += 256) { float v = xp[i]; s += v; ss += v * v; }
  __shared__ float red[16];
  for (int off = 16; off; off >>= 1) { s += SHFLX(s, off); ss += SHFLX(ss, off); }
  if (lane == 0) { red[w] = s; red[8 + w] = ss; }
  __syncthreads();
  s = 0.f; ss = 0.f;
  #pragma unroll
  for (int j = 0; j < 8; ++j) { s += red[j]; ss += red[8 + j]; }
  const float mean = s * (1.f / 16384.f);
  const float var = ss * (1.f / 16384.f) - mean * mean;
  const float inv = rsqrtf(var + 1e-5f);
  for (int i = tid; i < 16 * LL; i += 256) {
    int c = g * 16 + (i >> 10);
    float v = (xp[i] - mean) * inv * gw[c] + gb[c];
    outH[base + i] = (_Float16)v;
  }
}

// ================= LayerNorm(cond) -> transposed f16 [B][768][80] ============
__global__ __launch_bounds__(256) void layernorm_kernel(
    const float* __restrict__ cond, const float* __restrict__ lw,
    const float* __restrict__ lb, _Float16* __restrict__ cnT) {
  const int t = blockIdx.x % LC, n = blockIdx.x / LC;
  const int tid = threadIdx.x, lane = tid & 31, w = tid >> 5;
  const float* xp = cond + ((size_t)n * LC + t) * DD;
  float s = 0.f, ss = 0.f;
  for (int i = tid; i < DD; i += 256) { float v = xp[i]; s += v; ss += v * v; }
  __shared__ float red[16];
  for (int off = 16; off; off >>= 1) { s += SHFLX(s, off); ss += SHFLX(ss, off); }
  if (lane == 0) { red[w] = s; red[8 + w] = ss; }
  __syncthreads();
  s = 0.f; ss = 0.f;
  #pragma unroll
  for (int j = 0; j < 8; ++j) { s += red[j]; ss += red[8 + j]; }
  const float mean = s / (float)DD;
  const float var = ss / (float)DD - mean * mean;
  const float inv = rsqrtf(var + 1e-5f);
  for (int i = tid; i < DD; i += 256) {
    float v = (xp[i] - mean) * inv * lw[i] + lb[i];
    cnT[((size_t)n * DD + i) * LCP + t] = (_Float16)v;
  }
}

// ================= generic WMMA GEMM =================
// Out[z][M][ld] = W[M,K] (fp32->f16) @ X (f16) + bias; optional GELU,
// optional fp32 residual, writes f32 and/or f16.
// NMAJOR=false: X is [z][K][ld] (K-major rows) -> transpose-stage into LDS.
// NMAJOR=true : X is [z][ld][K] (N-major rows) -> straight-copy stage.
template <int BN, int WMT, int WNT, bool NMAJOR>
__global__ __launch_bounds__(256) void gemm_wmma_kernel(
    const float* __restrict__ Wm, const _Float16* __restrict__ X,
    const float* __restrict__ bias, float* __restrict__ outF,
    _Float16* __restrict__ outH, const float* __restrict__ resF,
    int M, int K, int ld, int act) {
  constexpr int BM = 128, BK = 32;
  constexpr int WAVES_N = BN / (16 * WNT);
  __shared__ _Float16 As[BM * BK];   // row-major [m][k]
  __shared__ _Float16 BsT[BN * BK];  // N-major  [n][k]
  const int tid = threadIdx.x, lane = tid & 31, w = tid >> 5;
  const int wm = w / WAVES_N, wn = w % WAVES_N;
  const int bm0 = blockIdx.y * BM, bn0 = blockIdx.x * BN;
  const size_t z = blockIdx.z;
  const _Float16* Xb = X + z * (size_t)K * ld;
  v8f acc[WMT][WNT] = {};

  for (int k0 = 0; k0 < K; k0 += BK) {
    { // A: 128x32 fp32 -> f16 ; 16 floats/thread
      const int idx = tid * 16;
      const int row = idx >> 5, col = idx & 31;
      const float* g = Wm + (size_t)(bm0 + row) * K + k0 + col;
      _Float16* sdst = As + row * BK + col;
      #pragma unroll
      for (int vv = 0; vv < 4; ++vv) {
        v4f f = *(const v4f*)(g + vv * 4);
        sdst[vv * 4 + 0] = (_Float16)f[0];
        sdst[vv * 4 + 1] = (_Float16)f[1];
        sdst[vv * 4 + 2] = (_Float16)f[2];
        sdst[vv * 4 + 3] = (_Float16)f[3];
      }
    }
    if constexpr (NMAJOR) {  // B: straight copy rows [n][k]
      constexpr int NV = (BK * BN) / 8;
      for (int i = tid; i < NV; i += 256) {
        const int n = i / (BK / 8), kc = (i % (BK / 8)) * 8;
        *(v8h*)(BsT + n * BK + kc) =
            *(const v8h*)(Xb + (size_t)(bn0 + n) * K + k0 + kc);
      }
    } else {  // B: BKxBN f16 global (K-major) -> transposed [n][k] in LDS
      constexpr int NV = (BK * BN) / 8;
      for (int i = tid; i < NV; i += 256) {
        const int kr = i / (BN / 8), nc = (i % (BN / 8)) * 8;
        v8h xv = *(const v8h*)(Xb + (size_t)(k0 + kr) * ld + bn0 + nc);
        #pragma unroll
        for (int j = 0; j < 8; ++j) BsT[(nc + j) * BK + kr] = xv[j];
      }
    }
    __syncthreads();
    v16h af[WMT], bf[WNT];
    #pragma unroll
    for (int mt = 0; mt < WMT; ++mt)
      af[mt] = frag_a_rm(As + (size_t)(wm * WMT + mt) * 16 * BK, BK, lane);
    #pragma unroll
    for (int nt = 0; nt < WNT; ++nt)
      bf[nt] = frag_b_nm(BsT + (size_t)(wn * WNT + nt) * 16 * BK, BK, lane);
    #pragma unroll
    for (int mt = 0; mt < WMT; ++mt)
      #pragma unroll
      for (int nt = 0; nt < WNT; ++nt)
        acc[mt][nt] = wmma16(af[mt], bf[nt], acc[mt][nt]);
    __syncthreads();
  }
  // epilogue (out always [z][M][ld] row-major; lanes contiguous in n)
  const int mlo = lane & 15, mhi = lane >> 4;
  #pragma unroll
  for (int mt = 0; mt < WMT; ++mt)
    #pragma unroll
    for (int nt = 0; nt < WNT; ++nt)
      #pragma unroll
      for (int r = 0; r < 8; ++r) {
        const int gm = bm0 + (wm * WMT + mt) * 16 + r + 8 * mhi;
        const int gn = bn0 + (wn * WNT + nt) * 16 + mlo;
        float v = acc[mt][nt][r] + bias[gm];
        if (act == 1) v = gelu_exact(v);
        const size_t o = z * (size_t)M * ld + (size_t)gm * ld + gn;
        if (resF) v += resF[o];
        if (outF) outF[o] = v;
        if (outH) outH[o] = (_Float16)v;
      }
}

// ================= self-attention (flash, per (b,h), 128-query tiles) ========
// output token-major: hout[B][L][C]  (coalesced: lanes contiguous in c)
__global__ __launch_bounds__(256) void attn_self_kernel(
    const _Float16* __restrict__ qkv,  // [B][1536][1024]
    float* __restrict__ hout) {        // [B][1024][512] token-major
  __shared__ _Float16 Qs[128 * 64];    // [t][c]  (A, row-major)
  __shared__ _Float16 KsT[64 * 64];    // [s][c]  (B N-major: N=s, K=c)
  __shared__ _Float16 Vs[64 * 64];     // [c][s]  (B N-major: N=c, K=s)
  __shared__ _Float16 Ps[8 * 16 * 64]; // per-wave P [16t][64s]
  const int tid = threadIdx.x, lane = tid & 31, w = tid >> 5;
  const int bh = blockIdx.y, b = bh >> 3, h = bh & 7;
  const int t0 = blockIdx.x * 128;
  const _Float16* qb = qkv + ((size_t)b * HPQ + h * CHD) * LL;
  const _Float16* kb = qb + (size_t)CC * LL;
  const _Float16* vb = kb + (size_t)CC * LL;
  // load Q transposed into [t][c]
  for (int i = tid; i < 1024; i += 256) {
    const int c = i >> 4, tt = (i & 15) * 8;
    v8h qv = *(const v8h*)(qb + (size_t)c * LL + t0 + tt);
    #pragma unroll
    for (int j = 0; j < 8; ++j) Qs[(tt + j) * 64 + c] = qv[j];
  }
  v8f acc_o[4] = {};   // out tiles [t=16][c=16] x4 over head dim
  float m_st[8], l_st[8];
  #pragma unroll
  for (int r = 0; r < 8; ++r) { m_st[r] = -3.0e38f; l_st[r] = 0.f; }

  for (int s0 = 0; s0 < LL; s0 += 64) {
    for (int i = tid; i < 512; i += 256) {  // K (transposed) and V tiles
      const int c = i >> 3, sc = (i & 7) * 8;
      v8h kv = *(const v8h*)(kb + (size_t)c * LL + s0 + sc);
      #pragma unroll
      for (int j = 0; j < 8; ++j) KsT[(sc + j) * 64 + c] = kv[j];
      *(v8h*)(Vs + c * 64 + sc) = *(const v8h*)(vb + (size_t)c * LL + s0 + sc);
    }
    __syncthreads();
    // S = (Q^T K) * 1/8  for this wave's 16 t rows
    v8f sacc[4] = {};
    #pragma unroll
    for (int ks = 0; ks < 64; ks += 32) {
      v16h a = frag_a_rm(Qs + (size_t)(w * 16) * 64 + ks, 64, lane);
      #pragma unroll
      for (int nt = 0; nt < 4; ++nt) {
        v16h bfr = frag_b_nm(KsT + (size_t)(nt * 16) * 64 + ks, 64, lane);
        sacc[nt] = wmma16(a, bfr, sacc[nt]);
      }
    }
    // online softmax (rows live across 16-lane halves)
    #pragma unroll
    for (int r = 0; r < 8; ++r) {
      float mx = -3.0e38f;
      #pragma unroll
      for (int nt = 0; nt < 4; ++nt) {
        sacc[nt][r] *= 0.125f;
        mx = fmaxf(mx, sacc[nt][r]);
      }
      for (int off = 1; off < 16; off <<= 1) mx = fmaxf(mx, SHFLX(mx, off));
      const float mnew = fmaxf(m_st[r], mx);
      const float corr = __expf(m_st[r] - mnew);
      float rs = 0.f;
      #pragma unroll
      for (int nt = 0; nt < 4; ++nt) {
        float p = __expf(sacc[nt][r] - mnew);
        sacc[nt][r] = p; rs += p;
      }
      for (int off = 1; off < 16; off <<= 1) rs += SHFLX(rs, off);
      l_st[r] = l_st[r] * corr + rs;
      m_st[r] = mnew;
      #pragma unroll
      for (int ct = 0; ct < 4; ++ct) acc_o[ct][r] *= corr;
      const int m = r + 8 * (lane >> 4);
      #pragma unroll
      for (int nt = 0; nt < 4; ++nt)
        Ps[(w * 16 + m) * 64 + nt * 16 + (lane & 15)] = (_Float16)sacc[nt][r];
    }
    // O += P @ V^T  (A = P [t][s], B = V^T via N-major V [c][s])
    #pragma unroll
    for (int ks = 0; ks < 64; ks += 32) {
      v16h a = frag_a_rm(Ps + (size_t)(w * 16) * 64 + ks, 64, lane);
      #pragma unroll
      for (int ct = 0; ct < 4; ++ct) {
        v16h bfr = frag_b_nm(Vs + (size_t)(ct * 16) * 64 + ks, 64, lane);
        acc_o[ct] = wmma16(a, bfr, acc_o[ct]);
      }
    }
    __syncthreads();
  }
  // token-major store: [b][t][c]; lanes 0..15 are contiguous in c
  #pragma unroll
  for (int ct = 0; ct < 4; ++ct)
    #pragma unroll
    for (int r = 0; r < 8; ++r) {
      const int m = r + 8 * (lane >> 4);
      const int c = h * CHD + ct * 16 + (lane & 15);
      hout[((size_t)b * LL + t0 + w * 16 + m) * CC + c] = acc_o[ct][r] / l_st[r];
    }
}

// ================= cross-attention (77 keys, single-pass softmax) ============
// hself / outH are token-major [B][L][C]
__global__ __launch_bounds__(256) void attn_cross_kernel(
    const _Float16* __restrict__ qkv,   // [B][1536][1024] (q rows 0..511)
    const _Float16* __restrict__ kvc,   // [B][1024][80] (kc 0..511, vc 512..)
    const int* __restrict__ mask,       // [B][77]
    const float* __restrict__ hself,    // [B][1024][512]
    _Float16* __restrict__ outH) {      // [B][1024][512] f16(hself + cross)
  __shared__ _Float16 Qs[128 * 64];     // [t][c]
  __shared__ _Float16 KcT[80 * 64];     // [s][c]
  __shared__ _Float16 Vc[64 * 96];      // [c][s] padded s to 96
  __shared__ _Float16 Ps[8 * 16 * 96];  // per-wave P, padded
  __shared__ float addm[LCP];
  const int tid = threadIdx.x, lane = tid & 31, w = tid >> 5;
  const int bh = blockIdx.y, b = bh >> 3, h = bh & 7;
  const int t0 = blockIdx.x * 128;
  const _Float16* qb = qkv + ((size_t)b * HPQ + h * CHD) * LL;
  const _Float16* kb = kvc + ((size_t)b * 1024 + h * CHD) * LCP;
  const _Float16* vb = kb + (size_t)CC * LCP;
  for (int i = tid; i < 64 * 96; i += 256) Vc[i] = (_Float16)0.f;
  for (int i = tid; i < 8 * 16 * 96; i += 256) Ps[i] = (_Float16)0.f;
  if (tid < LCP)
    addm[tid] = (tid < LC && mask[b * LC + tid] != 0) ? 0.f : -1e30f;
  for (int i = tid; i < 1024; i += 256) {
    const int c = i >> 4, tt = (i & 15) * 8;
    v8h qv = *(const v8h*)(qb + (size_t)c * LL + t0 + tt);
    #pragma unroll
    for (int j = 0; j < 8; ++j) Qs[(tt + j) * 64 + c] = qv[j];
  }
  __syncthreads();  // zero fill complete before partial overwrite
  for (int i = tid; i < 640; i += 256) {  // 64 rows x 80 cols, v8h granules
    const int c = i / 10, sc = (i % 10) * 8;
    v8h kv = *(const v8h*)(kb + (size_t)c * LCP + sc);
    #pragma unroll
    for (int j = 0; j < 8; ++j) KcT[(sc + j) * 64 + c] = kv[j];
    *(v8h*)(Vc + c * 96 + sc) = *(const v8h*)(vb + (size_t)c * LCP + sc);
  }
  __syncthreads();
  // S = Q^T Kc
  v8f sacc[5] = {};
  #pragma unroll
  for (int ks = 0; ks < 64; ks += 32) {
    v16h a = frag_a_rm(Qs + (size_t)(w * 16) * 64 + ks, 64, lane);
    #pragma unroll
    for (int nt = 0; nt < 5; ++nt) {
      v16h bfr = frag_b_nm(KcT + (size_t)(nt * 16) * 64 + ks, 64, lane);
      sacc[nt] = wmma16(a, bfr, sacc[nt]);
    }
  }
  float l_loc[8];
  #pragma unroll
  for (int r = 0; r < 8; ++r) {
    float mx = -3.0e38f;
    #pragma unroll
    for (int nt = 0; nt < 5; ++nt) {
      float lg = sacc[nt][r] * 0.125f + addm[nt * 16 + (lane & 15)];
      sacc[nt][r] = lg;
      mx = fmaxf(mx, lg);
    }
    for (int off = 1; off < 16; off <<= 1) mx = fmaxf(mx, SHFLX(mx, off));
    float rs = 0.f;
    #pragma unroll
    for (int nt = 0; nt < 5; ++nt) {
      float p = __expf(sacc[nt][r] - mx);
      sacc[nt][r] = p; rs += p;
    }
    for (int off = 1; off < 16; off <<= 1) rs += SHFLX(rs, off);
    l_loc[r] = rs;
    const int m = r + 8 * (lane >> 4);
    #pragma unroll
    for (int nt = 0; nt < 5; ++nt)
      Ps[(w * 16 + m) * 96 + nt * 16 + (lane & 15)] = (_Float16)sacc[nt][r];
  }
  v8f oacc[4] = {};
  #pragma unroll
  for (int ks = 0; ks < 96; ks += 32) {
    v16h a = frag_a_rm(Ps + (size_t)(w * 16) * 96 + ks, 96, lane);
    #pragma unroll
    for (int ct = 0; ct < 4; ++ct) {
      v16h bfr = frag_b_nm(Vc + (size_t)(ct * 16) * 96 + ks, 96, lane);
      oacc[ct] = wmma16(a, bfr, oacc[ct]);
    }
  }
  #pragma unroll
  for (int ct = 0; ct < 4; ++ct)
    #pragma unroll
    for (int r = 0; r < 8; ++r) {
      const int m = r + 8 * (lane >> 4);
      const int c = h * CHD + ct * 16 + (lane & 15);
      const size_t o = ((size_t)b * LL + t0 + w * 16 + m) * CC + c;
      outH[o] = (_Float16)(hself[o] + oacc[ct][r] / l_loc[r]);
    }
}

// ================= host-side launch =================
extern "C" void kernel_launch(void* const* d_in, const int* in_sizes, int n_in,
                              void* d_out, int out_size, void* d_ws,
                              size_t ws_size, hipStream_t stream) {
  const float* x        = (const float*)d_in[0];
  const float* cond     = (const float*)d_in[1];
  const int*   cmask    = (const int*)d_in[2];
  const float* gn_w     = (const float*)d_in[3];
  const float* gn_b     = (const float*)d_in[4];
  const float* qkv_w    = (const float*)d_in[5];
  const float* qkv_b    = (const float*)d_in[6];
  const float* ln_w     = (const float*)d_in[7];
  const float* ln_b     = (const float*)d_in[8];
  const float* kvc_w    = (const float*)d_in[9];
  const float* kvc_b    = (const float*)d_in[10];
  const float* proj_w   = (const float*)d_in[11];
  const float* proj_b   = (const float*)d_in[12];
  const float* ffn_gn_w = (const float*)d_in[13];
  const float* ffn_gn_b = (const float*)d_in[14];
  const float* ffn_w1   = (const float*)d_in[15];
  const float* ffn_b1   = (const float*)d_in[16];
  const float* ffn_w2   = (const float*)d_in[17];
  const float* ffn_b2   = (const float*)d_in[18];
  float* out = (float*)d_out;

  const size_t MB = 1024ull * 1024ull;
  char* ws = (char*)d_ws;
  _Float16* qkv_h = (_Float16*)(ws + 0);        // 24 MB [B][1536][1024]
  _Float16* xn_h  = (_Float16*)(ws + 24 * MB);  //  8 MB [B][512][1024]
  float*    h_f32 = (float*)(ws + 32 * MB);     // 16 MB [B][1024][512] tok-major
  _Float16* h16   = (_Float16*)(ws + 48 * MB);  //  8 MB [B][1024][512] tok-major
  float*    x1    = (float*)(ws + 56 * MB);     // 16 MB [B][512][1024]
  _Float16* g2_h  = (_Float16*)(ws + 72 * MB);  //  8 MB
  _Float16* cnT   = (_Float16*)(ws + 80 * MB);  // ~1 MB [B][768][80]
  _Float16* kvc_h = (_Float16*)(ws + 82 * MB);  // ~1.3MB [B][1024][80]
  _Float16* y_h   = (_Float16*)(ws + 0);        // 32 MB, aliases qkv_h+xn_h (dead)

  // 0) zero cnT padding columns
  {
    int n = BQ * DD * LCP;
    zero_f16_kernel<<<(n + 255) / 256, 256, 0, stream>>>(cnT, n);
  }
  // 1) GroupNorm 1 -> f16
  groupnorm_kernel<<<BQ * 32, 256, 0, stream>>>(x, gn_w, gn_b, xn_h);
  // 2) QKV GEMM: [1536,512] @ [512,1024]
  gemm_wmma_kernel<128, 4, 2, false>
      <<<dim3(LL / 128, HPQ / 128, BQ), 256, 0, stream>>>(
          qkv_w, xn_h, qkv_b, nullptr, qkv_h, nullptr, HPQ, CC, LL, 0);
  // 3) LayerNorm(cond) -> transposed f16
  layernorm_kernel<<<BQ * LC, 256, 0, stream>>>(cond, ln_w, ln_b, cnT);
  // 4) KVC GEMM: [1024,768] @ [768,80]
  gemm_wmma_kernel<16, 1, 1, false>
      <<<dim3(LCP / 16, 1024 / 128, BQ), 256, 0, stream>>>(
          kvc_w, cnT, kvc_b, nullptr, kvc_h, nullptr, 1024, DD, LCP, 0);
  // 5) self-attention (flash) -> h_f32 (token-major)
  attn_self_kernel<<<dim3(LL / 128, BQ * NH), 256, 0, stream>>>(qkv_h, h_f32);
  // 6) cross-attention, add + pack f16 (token-major)
  attn_cross_kernel<<<dim3(LL / 128, BQ * NH), 256, 0, stream>>>(
      qkv_h, kvc_h, cmask, h_f32, h16);
  // 7) proj GEMM (+x residual) with token-major (NMAJOR) input -> x1 (f32)
  gemm_wmma_kernel<128, 4, 2, true>
      <<<dim3(LL / 128, CC / 128, BQ), 256, 0, stream>>>(
          proj_w, h16, proj_b, x1, nullptr, x, CC, CC, LL, 0);
  // 8) GroupNorm 2 -> f16
  groupnorm_kernel<<<BQ * 32, 256, 0, stream>>>(x1, ffn_gn_w, ffn_gn_b, g2_h);
  // 9) FFN1 GEMM + exact GELU -> f16
  gemm_wmma_kernel<128, 4, 2, false>
      <<<dim3(LL / 128, 2048 / 128, BQ), 256, 0, stream>>>(
          ffn_w1, g2_h, ffn_b1, nullptr, y_h, nullptr, 2048, CC, LL, 1);
  // 10) FFN2 GEMM + x1 residual -> d_out (f32)
  gemm_wmma_kernel<128, 4, 2, false>
      <<<dim3(LL / 128, CC / 128, BQ), 256, 0, stream>>>(
          ffn_w2, y_h, ffn_b2, out, nullptr, x1, CC, 2048, LL, 0);
  (void)in_sizes; (void)n_in; (void)out_size; (void)ws_size;
}
